// MMDiTBlock_40303973105880
// MI455X (gfx1250) — matrix operand order repaired and checked by reading, not compiled, not run on hardware
//
#include <hip/hip_runtime.h>
#include <hip/hip_bf16.h>
#include <stdint.h>

#define DEV static __device__ __forceinline__

typedef __attribute__((ext_vector_type(16))) __bf16    v16bf;
typedef __attribute__((ext_vector_type(8)))  float     v8f;
typedef __attribute__((ext_vector_type(4)))  uint32_t  u32x4;
typedef __attribute__((ext_vector_type(8)))  int       i32x8;
typedef __attribute__((ext_vector_type(4)))  int       i32x4;

struct U32x8 { uint4 lo, hi; };

DEV uint16_t f2bf(float f) {
  uint32_t u = __builtin_bit_cast(uint32_t, f);
  u += 0x7FFFu + ((u >> 16) & 1u);
  return (uint16_t)(u >> 16);
}

DEV v16bf ldfrag(const uint16_t* plo, const uint16_t* phi) {
  U32x8 t;
  t.lo = *(const uint4*)plo;
  t.hi = *(const uint4*)phi;
  return __builtin_bit_cast(v16bf, t);
}

DEV v8f wmma_bf16(v16bf a, v16bf b, v8f c) {
  return __builtin_amdgcn_wmma_f32_16x16x32_bf16(false, a, false, b, (short)0, c, false, false);
}

DEV v8f z8() { v8f z = {0.f,0.f,0.f,0.f,0.f,0.f,0.f,0.f}; return z; }

// block = 256 threads (8 waves); reduce sum across block
DEV float blk_reduce_sum(float v, float* red) {
  #pragma unroll
  for (int d = 16; d; d >>= 1) v += __shfl_xor(v, d);
  if ((threadIdx.x & 31) == 0) red[threadIdx.x >> 5] = v;
  __syncthreads();
  float r = 0.f;
  #pragma unroll
  for (int i = 0; i < 8; ++i) r += red[i];
  __syncthreads();
  return r;
}

// ---------------- TDM: DMA a 128x32 bf16 tile (row stride K elems) to LDS --
// D# per CDNA5 ISA.
// pad: 4 DWORDs after every 16 DWORDs (64B row) -> LDS row stride 80B (40 u16)
DEV void tdm_load_tile(uint32_t ldsOff, const void* gaddr, int K) {
  uint64_t ga = (uint64_t)(uintptr_t)gaddr;
  u32x4 g0;
  g0[0] = 1u;                                            // count=1, user mode
  g0[1] = ldsOff;                                        // lds_addr (bytes)
  g0[2] = (uint32_t)ga;                                  // global_addr[31:0]
  g0[3] = (uint32_t)((ga >> 32) & 0x01FFFFFFu) | (2u << 30); // addr[56:32], type=2
  i32x8 g1;
  g1[0] = (1 << 16) | (1 << 20) | (3 << 22) | (3 << 25); // data 2B; pad_en; ivl=16DW; amt=4DW
  g1[1] = (int)(((uint32_t)K & 0xFFFFu) << 16);          // tensor_dim0 lo16
  g1[2] = (int)((uint32_t)K >> 16);                      // tensor_dim0 hi16, dim1 lo16=0
  g1[3] = (int)(16u | (32u << 16));                      // dim1 hi16 (=1M rows), tile_dim0=32
  g1[4] = 128;                                           // tile_dim1=128, tile_dim2=0
  g1[5] = K;                                             // tensor_dim0_stride lo32
  g1[6] = 0;                                             // stride hi16, dim1_stride lo=0
  g1[7] = 0;
  i32x4 z4 = {0, 0, 0, 0};
#if defined(__clang_major__) && (__clang_major__ >= 23)
  i32x8 z8i = {0, 0, 0, 0, 0, 0, 0, 0};
  __builtin_amdgcn_tensor_load_to_lds(g0, g1, z4, z4, z8i, 0);
#else
  __builtin_amdgcn_tensor_load_to_lds(g0, g1, z4, z4, 0);
#endif
}

// ---------------- weight convert: f32 [K][N] -> bf16 [N][K] ----------------
__global__ __launch_bounds__(256)
void wconv_kernel(const float* __restrict__ W, uint16_t* __restrict__ Wt, int K, int N) {
  __shared__ float tile[32][33];
  const int bn = blockIdx.x * 32, bk = blockIdx.y * 32;
  const int tx = threadIdx.x & 31, ty = threadIdx.x >> 5;
  for (int r = ty; r < 32; r += 8)
    tile[r][tx] = W[(size_t)(bk + r) * N + bn + tx];
  __syncthreads();
  for (int r = ty; r < 32; r += 8)
    Wt[(size_t)(bn + r) * K + bk + tx] = f2bf(tile[tx][r]);
}

// ---------------- modulation: silu(y) @ mod_W + b  (tiny GEMV) -------------
__global__ __launch_bounds__(256)
void mod_kernel(const float* __restrict__ y,
                const float* __restrict__ Wi, const float* __restrict__ bi,
                const float* __restrict__ Wc, const float* __restrict__ bc,
                float* __restrict__ mod_i, float* __restrict__ mod_c) {
  __shared__ float ya[1024];
  const int b = blockIdx.x;
  const int j = blockIdx.y * 256 + threadIdx.x;   // 0..12287
  for (int k = threadIdx.x; k < 1024; k += 256) {
    float v = y[b * 1024 + k];
    ya[k] = v / (1.f + expf(-v));
  }
  __syncthreads();
  const bool isCtx = j >= 6144;
  const int col = isCtx ? j - 6144 : j;
  const float* W = isCtx ? Wc : Wi;
  float s = isCtx ? bc[col] : bi[col];
  for (int k = 0; k < 1024; ++k) s += ya[k] * W[(size_t)k * 6144 + col];
  (isCtx ? mod_c : mod_i)[b * 6144 + col] = s;
}

// ---------------- RMS-norm + (1+scale)*x + shift -> bf16 -------------------
__global__ __launch_bounds__(256)
void modnorm_kernel(const float* __restrict__ src, const float* __restrict__ nw,
                    const float* __restrict__ mod, int shIdx, int scIdx,
                    uint16_t* __restrict__ out, int tpb) {
  __shared__ float red[8];
  const int row = blockIdx.x;
  const int b = row / tpb;
  const float* x = src + (size_t)row * 1024;
  float v[4]; float ss = 0.f;
  #pragma unroll
  for (int i = 0; i < 4; ++i) { v[i] = x[threadIdx.x + i * 256]; ss += v[i] * v[i]; }
  float tot = blk_reduce_sum(ss, red);
  float rs = rsqrtf(tot * (1.f / 1024.f) + 1e-6f);
  const float* mb = mod + (size_t)b * 6144;
  #pragma unroll
  for (int i = 0; i < 4; ++i) {
    int c = threadIdx.x + i * 256;
    float o = v[i] * rs * nw[c] * (1.f + mb[scIdx * 1024 + c]) + mb[shIdx * 1024 + c];
    out[(size_t)row * 1024 + c] = f2bf(o);
  }
}

// ---------------- WMMA GEMM: C[M,N] = A[M,K](bf16) @ Wt[N,K](bf16) + bias --
// Tiles DMA'd by the Tensor Data Mover into double-buffered LDS
// (dynamic LDS only => TDM lds_addr offsets are the literal byte offsets).
enum { EPI_F32 = 0, EPI_GELU_BF16 = 1, EPI_GATE = 2 };

template <int EPI>
__global__ __launch_bounds__(256)
void gemm_bf16(const uint16_t* __restrict__ A, const uint16_t* __restrict__ Wt,
               const float* __restrict__ bias,
               float* __restrict__ outF, uint16_t* __restrict__ outH,
               const float* __restrict__ resid, const float* __restrict__ gate,
               int M, int N, int K, int tpb) {
  extern __shared__ uint16_t ldsbuf[];        // 4 x 10240B: A0 B0 A1 B1
  const int m0 = blockIdx.x * 128, n0 = blockIdx.y * 128;
  const int t = threadIdx.x;
  const int wid = t >> 5, lane = t & 31;
  const int wr = wid >> 2, wc = wid & 3;      // 2x4 wave grid
  const int lhi = lane >> 4, lo = lane & 15;
  const bool issuer = (wid == 0);
  (void)M;

  v8f acc[4][2];
  #pragma unroll
  for (int i = 0; i < 4; ++i)
    #pragma unroll
    for (int j = 0; j < 2; ++j) acc[i][j] = z8();

  if (issuer) {
    tdm_load_tile(0u,     A  + (size_t)m0 * K, K);
    tdm_load_tile(10240u, Wt + (size_t)n0 * K, K);
  }

  const int nk = K >> 5;
  for (int kt = 0; kt < nk; ++kt) {
    const int cur = kt & 1;
    if (issuer) __builtin_amdgcn_s_wait_tensorcnt(0);
    __syncthreads();                          // hand DMA'd tiles to all waves
    if (issuer && (kt + 1 < nk)) {
      const int k0 = (kt + 1) << 5;
      const uint32_t nxtA = (kt & 1) ? 0u : 20480u;
      tdm_load_tile(nxtA,          A  + (size_t)m0 * K + k0, K);
      tdm_load_tile(nxtA + 10240u, Wt + (size_t)n0 * K + k0, K);
    }
    const uint16_t* As = ldsbuf + (cur ? 10240 : 0);
    const uint16_t* Bs = ldsbuf + (cur ? 15360 : 5120);

    v16bf af[4], bfr[2];
    #pragma unroll
    for (int i = 0; i < 4; ++i) {
      const uint16_t* p = &As[(wr * 64 + i * 16 + lo) * 40];
      af[i] = ldfrag(p + lhi * 8, p + 16 + lhi * 8);
    }
    #pragma unroll
    for (int j = 0; j < 2; ++j) {
      const uint16_t* p = &Bs[(wc * 32 + j * 16 + lo) * 40];
      bfr[j] = ldfrag(p + lhi * 8, p + 16 + lhi * 8);
    }
    #pragma unroll
    for (int i = 0; i < 4; ++i)
      #pragma unroll
      for (int j = 0; j < 2; ++j)
        acc[i][j] = wmma_bf16(af[i], bfr[j], acc[i][j]);
    __syncthreads();                          // compute done before next DMA lands
  }

  #pragma unroll
  for (int i = 0; i < 4; ++i) {
    int mBase = m0 + wr * 64 + i * 16 + lhi * 8;
    #pragma unroll
    for (int j = 0; j < 2; ++j) {
      int n = n0 + wc * 32 + j * 16 + lo;
      float bv = bias[n];
      #pragma unroll
      for (int v = 0; v < 8; ++v) {
        int m = mBase + v;
        float val = acc[i][j][v] + bv;
        size_t off = (size_t)m * N + n;
        if (EPI == EPI_F32) {
          outF[off] = val;
        } else if (EPI == EPI_GELU_BF16) {
          float g = 0.5f * val * (1.f + tanhf(0.7978845608028654f * (val + 0.044715f * val * val * val)));
          outH[off] = f2bf(g);
        } else {
          int bb = m / tpb;
          outF[off] = resid[off] + gate[(size_t)bb * 6144 + n] * val;
        }
      }
    }
  }
}

// ---------------- QK rms-norm + RoPE + scatter to [b][h][pos][64] ----------
__global__ __launch_bounds__(256)
void qkpost_kernel(const float* __restrict__ qkv, const float* __restrict__ qnw,
                   const float* __restrict__ knw,
                   uint16_t* __restrict__ qb, uint16_t* __restrict__ kb, uint16_t* __restrict__ vb,
                   int tpb, int posOff, int rope2d, const int* __restrict__ Wptr, float base) {
  __shared__ float qs[1024];
  __shared__ float ks[1024];
  __shared__ float red[8];
  const int row = blockIdx.x;
  const int b = row / tpb;
  const int s = row - b * tpb;
  const int pos = posOff + s;
  const float* r0 = qkv + (size_t)row * 3072;
  float sq = 0.f, sk = 0.f;
  for (int i = threadIdx.x; i < 1024; i += 256) {
    float q = r0[i], k = r0[1024 + i], v = r0[2048 + i];
    qs[i] = q; ks[i] = k;
    sq += q * q; sk += k * k;
    int hh = i >> 6, dd = i & 63;
    vb[((size_t)(b * 16 + hh) * 1280 + pos) * 64 + dd] = f2bf(v);
  }
  float sqT = blk_reduce_sum(sq, red);
  float skT = blk_reduce_sum(sk, red);
  float rsq = rsqrtf(sqT * (1.f / 1024.f) + 1e-6f);
  float rsk = rsqrtf(skT * (1.f / 1024.f) + 1e-6f);
  const float l2b = log2f(base);
  float prow = 0.f, pcol = 0.f;
  if (rope2d) {
    int Wd = Wptr[0];
    prow = (float)(s / Wd);
    pcol = (float)(s - (s / Wd) * Wd);
  }
  for (int i = threadIdx.x; i < 1024; i += 256) {
    int hh = i >> 6, dd = i & 63;
    int i1, i2; float ang; bool second;
    if (!rope2d) {
      int j = dd & 31; second = dd >= 32;
      ang = (float)pos * exp2f(-l2b * (float)j * (1.f / 32.f));
      i1 = (hh << 6) + j; i2 = i1 + 32;
    } else {
      int half = (dd >> 5) & 1;
      int d2 = dd & 31;
      int j = d2 & 15; second = d2 >= 16;
      float p = half ? pcol : prow;
      ang = p * exp2f(-l2b * (float)j * (1.f / 16.f));
      i1 = (hh << 6) + (half << 5) + j; i2 = i1 + 16;
    }
    float sn, cs; sincosf(ang, &sn, &cs);
    float q1 = qs[i1] * rsq * qnw[i1], q2 = qs[i2] * rsq * qnw[i2];
    float k1 = ks[i1] * rsk * knw[i1], k2 = ks[i2] * rsk * knw[i2];
    float qv = second ? (q1 * sn + q2 * cs) : (q1 * cs - q2 * sn);
    float kv = second ? (k1 * sn + k2 * cs) : (k1 * cs - k2 * sn);
    size_t o = ((size_t)(b * 16 + hh) * 1280 + pos) * 64 + dd;
    qb[o] = f2bf(qv); kb[o] = f2bf(kv);
  }
}

// ---------------- Flash attention, WMMA bf16, S=1280, Dh=64 ----------------
__global__ __launch_bounds__(256)
void attn_kernel(const uint16_t* __restrict__ qb, const uint16_t* __restrict__ kb,
                 const uint16_t* __restrict__ vb, const unsigned char* __restrict__ mask,
                 uint16_t* __restrict__ outCtx, uint16_t* __restrict__ outIn) {
  __shared__ __align__(16) uint16_t Ks[32 * 72];
  __shared__ __align__(16) uint16_t Vt[64 * 40];
  __shared__ __align__(16) uint16_t Ps[8 * 640];
  const int bh = blockIdx.x;             // b*16+h
  const int qt = blockIdx.y;             // 0..9
  const int b = bh >> 4, h = bh & 15;
  const int t = threadIdx.x, wid = t >> 5, lane = t & 31;
  const int lhi = lane >> 4, lo = lane & 15;
  const size_t base = (size_t)bh * (1280 * 64);

  const int qrow = qt * 128 + wid * 16 + lo;
  const uint16_t* qp = qb + base + (size_t)qrow * 64;
  v16bf qa0 = ldfrag(qp + lhi * 8,      qp + 16 + lhi * 8);
  v16bf qa1 = ldfrag(qp + 32 + lhi * 8, qp + 48 + lhi * 8);

  v8f o[4];
  #pragma unroll
  for (int j = 0; j < 4; ++j) o[j] = z8();
  float mprev[8], l[8];
  #pragma unroll
  for (int v = 0; v < 8; ++v) { mprev[v] = -3.0e38f; l[v] = 0.f; }

  const int ldr = t >> 3, ldo = (t & 7) << 3;

  for (int kt = 0; kt < 40; ++kt) {
    {
      const uint16_t* kp = kb + base + (size_t)(kt * 32 + ldr) * 64 + ldo;
      *(uint4*)&Ks[ldr * 72 + ldo] = *(const uint4*)kp;
      const uint16_t* vp = vb + base + (size_t)(kt * 32 + ldr) * 64 + ldo;
      union { uint4 q; uint16_t s[8]; } u; u.q = *(const uint4*)vp;
      #pragma unroll
      for (int j = 0; j < 8; ++j) Vt[(ldo + j) * 40 + ldr] = u.s[j];
    }
    __syncthreads();

    v8f s0 = z8(), s1 = z8();
    #pragma unroll
    for (int kc = 0; kc < 2; ++kc) {
      const uint16_t* pb0 = &Ks[lo * 72 + kc * 32 + lhi * 8];
      const uint16_t* pb1 = &Ks[(16 + lo) * 72 + kc * 32 + lhi * 8];
      v16bf b0 = ldfrag(pb0, pb0 + 16);
      v16bf b1 = ldfrag(pb1, pb1 + 16);
      v16bf aq = kc ? qa1 : qa0;
      s0 = wmma_bf16(aq, b0, s0);
      s1 = wmma_bf16(aq, b1, s1);
    }
    const int c0 = kt * 32 + lo, c1 = c0 + 16;
    float add0 = (c0 < 256 && !mask[b * 256 + c0]) ? -1e30f : 0.f;
    float add1 = (c1 < 256 && !mask[b * 256 + c1]) ? -1e30f : 0.f;
    float e0[8], e1[8], rmax[8];
    #pragma unroll
    for (int v = 0; v < 8; ++v) {
      e0[v] = s0[v] * 0.125f + add0;
      e1[v] = s1[v] * 0.125f + add1;
      rmax[v] = fmaxf(e0[v], e1[v]);
    }
    #pragma unroll
    for (int d = 1; d < 16; d <<= 1)
      #pragma unroll
      for (int v = 0; v < 8; ++v)
        rmax[v] = fmaxf(rmax[v], __shfl_xor(rmax[v], d));
    float alpha[8], rsum[8];
    #pragma unroll
    for (int v = 0; v < 8; ++v) {
      float mn = fmaxf(mprev[v], rmax[v]);
      alpha[v] = __expf(mprev[v] - mn);
      mprev[v] = mn;
      e0[v] = __expf(e0[v] - mn);
      e1[v] = __expf(e1[v] - mn);
      rsum[v] = e0[v] + e1[v];
    }
    #pragma unroll
    for (int d = 1; d < 16; d <<= 1)
      #pragma unroll
      for (int v = 0; v < 8; ++v)
        rsum[v] += __shfl_xor(rsum[v], d);
    #pragma unroll
    for (int v = 0; v < 8; ++v) l[v] = l[v] * alpha[v] + rsum[v];
    #pragma unroll
    for (int j = 0; j < 4; ++j)
      #pragma unroll
      for (int v = 0; v < 8; ++v) o[j][v] *= alpha[v];

    uint16_t* P = &Ps[wid * 640];
    #pragma unroll
    for (int v = 0; v < 8; ++v) {
      int m = v + lhi * 8;
      P[m * 40 + lo] = f2bf(e0[v]);
      P[m * 40 + 16 + lo] = f2bf(e1[v]);
    }
    asm volatile("s_wait_dscnt 0" ::: "memory");   // wave-local LDS store->load
    const uint16_t* pp = &Ps[wid * 640 + lo * 40];
    v16bf pa = ldfrag(pp + lhi * 8, pp + 16 + lhi * 8);
    #pragma unroll
    for (int j = 0; j < 4; ++j) {
      const uint16_t* pv = &Vt[(j * 16 + lo) * 40 + lhi * 8];
      v16bf vf = ldfrag(pv, pv + 16);
      o[j] = wmma_bf16(pa, vf, o[j]);
    }
    __syncthreads();
  }

  #pragma unroll
  for (int v = 0; v < 8; ++v) {
    int srow = qt * 128 + wid * 16 + lhi * 8 + v;
    float li = 1.f / l[v];
    #pragma unroll
    for (int j = 0; j < 4; ++j) {
      int d = j * 16 + lo;
      float val = o[j][v] * li;
      if (srow < 256)
        outCtx[((size_t)(b * 256 + srow)) * 1024 + h * 64 + d] = f2bf(val);
      else
        outIn[((size_t)(b * 1024 + (srow - 256))) * 1024 + h * 64 + d] = f2bf(val);
    }
  }
}

// ---------------------------------------------------------------------------
extern "C" void kernel_launch(void* const* d_in, const int* in_sizes, int n_in,
                              void* d_out, int out_size, void* d_ws, size_t ws_size,
                              hipStream_t stream) {
  (void)in_sizes; (void)n_in; (void)out_size; (void)ws_size;
  const float* x          = (const float*)d_in[0];
  const float* y          = (const float*)d_in[1];
  const float* ctx        = (const float*)d_in[2];
  const unsigned char* am = (const unsigned char*)d_in[3];
  const int* Wp           = (const int*)d_in[5];
  const float* mod_in_W   = (const float*)d_in[6];
  const float* mod_in_b   = (const float*)d_in[7];
  const float* mod_ctx_W  = (const float*)d_in[8];
  const float* mod_ctx_b  = (const float*)d_in[9];
  const float* norm1_in   = (const float*)d_in[10];
  const float* norm1_ctx  = (const float*)d_in[11];
  const float* norm2_in   = (const float*)d_in[12];
  const float* norm2_ctx  = (const float*)d_in[13];
  const float* qkv_in_W   = (const float*)d_in[14];
  const float* qkv_in_b   = (const float*)d_in[15];
  const float* qkv_ctx_W  = (const float*)d_in[16];
  const float* qkv_ctx_b  = (const float*)d_in[17];
  const float* qn_in_q    = (const float*)d_in[18];
  const float* qn_in_k    = (const float*)d_in[19];
  const float* qn_ctx_q   = (const float*)d_in[20];
  const float* qn_ctx_k   = (const float*)d_in[21];
  const float* po_in_W    = (const float*)d_in[22];
  const float* po_in_b    = (const float*)d_in[23];
  const float* po_ctx_W   = (const float*)d_in[24];
  const float* po_ctx_b   = (const float*)d_in[25];
  const float* mlp_in_W1  = (const float*)d_in[26];
  const float* mlp_in_b1  = (const float*)d_in[27];
  const float* mlp_in_W2  = (const float*)d_in[28];
  const float* mlp_in_b2  = (const float*)d_in[29];
  const float* mlp_ctx_W1 = (const float*)d_in[30];
  const float* mlp_ctx_b1 = (const float*)d_in[31];
  const float* mlp_ctx_W2 = (const float*)d_in[32];
  const float* mlp_ctx_b2 = (const float*)d_in[33];

  float* outX = (float*)d_out;                                  // [4][1024][1024]
  float* outC = (float*)d_out + (size_t)4 * 1024 * 1024;        // [4][256][1024]

  char* wp_ = (char*)d_ws;
  auto alloc = [&](size_t bytes) -> void* {
    void* p = (void*)wp_;
    wp_ += (bytes + 255) & ~(size_t)255;
    return p;
  };

  uint16_t* wt_qkv_in  = (uint16_t*)alloc((size_t)3072 * 1024 * 2);
  uint16_t* wt_qkv_ctx = (uint16_t*)alloc((size_t)3072 * 1024 * 2);
  uint16_t* wt_po_in   = (uint16_t*)alloc((size_t)1024 * 1024 * 2);
  uint16_t* wt_po_ctx  = (uint16_t*)alloc((size_t)1024 * 1024 * 2);
  uint16_t* wt_mlp_i1  = (uint16_t*)alloc((size_t)4096 * 1024 * 2);
  uint16_t* wt_mlp_i2  = (uint16_t*)alloc((size_t)4096 * 1024 * 2);
  uint16_t* wt_mlp_c1  = (uint16_t*)alloc((size_t)4096 * 1024 * 2);
  uint16_t* wt_mlp_c2  = (uint16_t*)alloc((size_t)4096 * 1024 * 2);
  float*    mod_i      = (float*)alloc((size_t)4 * 6144 * 4);
  float*    mod_c      = (float*)alloc((size_t)4 * 6144 * 4);
  uint16_t* xmod       = (uint16_t*)alloc((size_t)4096 * 1024 * 2);
  uint16_t* cmod       = (uint16_t*)alloc((size_t)1024 * 1024 * 2);
  float*    qkvI       = (float*)alloc((size_t)4096 * 3072 * 4);
  float*    qkvC       = (float*)alloc((size_t)1024 * 3072 * 4);
  uint16_t* qbuf       = (uint16_t*)alloc((size_t)4 * 16 * 1280 * 64 * 2);
  uint16_t* kbuf       = (uint16_t*)alloc((size_t)4 * 16 * 1280 * 64 * 2);
  uint16_t* vbuf       = (uint16_t*)alloc((size_t)4 * 16 * 1280 * 64 * 2);
  uint16_t* attnI      = (uint16_t*)alloc((size_t)4096 * 1024 * 2);
  uint16_t* attnC      = (uint16_t*)alloc((size_t)1024 * 1024 * 2);
  uint16_t* xm2        = (uint16_t*)alloc((size_t)4096 * 1024 * 2);
  uint16_t* cm2        = (uint16_t*)alloc((size_t)1024 * 1024 * 2);
  uint16_t* hIn        = (uint16_t*)alloc((size_t)4096 * 4096 * 2);
  uint16_t* hC         = (uint16_t*)alloc((size_t)1024 * 4096 * 2);

  // weight convert/transpose: f32 [K][N] -> bf16 [N][K]
  wconv_kernel<<<dim3(96, 32),  256, 0, stream>>>(qkv_in_W,   wt_qkv_in,  1024, 3072);
  wconv_kernel<<<dim3(96, 32),  256, 0, stream>>>(qkv_ctx_W,  wt_qkv_ctx, 1024, 3072);
  wconv_kernel<<<dim3(32, 32),  256, 0, stream>>>(po_in_W,    wt_po_in,   1024, 1024);
  wconv_kernel<<<dim3(32, 32),  256, 0, stream>>>(po_ctx_W,   wt_po_ctx,  1024, 1024);
  wconv_kernel<<<dim3(128, 32), 256, 0, stream>>>(mlp_in_W1,  wt_mlp_i1,  1024, 4096);
  wconv_kernel<<<dim3(32, 128), 256, 0, stream>>>(mlp_in_W2,  wt_mlp_i2,  4096, 1024);
  wconv_kernel<<<dim3(128, 32), 256, 0, stream>>>(mlp_ctx_W1, wt_mlp_c1,  1024, 4096);
  wconv_kernel<<<dim3(32, 128), 256, 0, stream>>>(mlp_ctx_W2, wt_mlp_c2,  4096, 1024);

  mod_kernel<<<dim3(4, 48), 256, 0, stream>>>(y, mod_in_W, mod_in_b, mod_ctx_W, mod_ctx_b, mod_i, mod_c);

  modnorm_kernel<<<4096, 256, 0, stream>>>(x,   norm1_in,  mod_i, 0, 1, xmod, 1024);
  modnorm_kernel<<<1024, 256, 0, stream>>>(ctx, norm1_ctx, mod_c, 0, 1, cmod, 256);

  const size_t GEMM_LDS = 40960;   // 4 x (128x32 bf16 tile, padded to 80B rows)
  gemm_bf16<EPI_F32><<<dim3(32, 24), 256, GEMM_LDS, stream>>>(xmod, wt_qkv_in,  qkv_in_b,  qkvI, nullptr, nullptr, nullptr, 4096, 3072, 1024, 1024);
  gemm_bf16<EPI_F32><<<dim3(8,  24), 256, GEMM_LDS, stream>>>(cmod, wt_qkv_ctx, qkv_ctx_b, qkvC, nullptr, nullptr, nullptr, 1024, 3072, 1024, 256);

  qkpost_kernel<<<1024, 256, 0, stream>>>(qkvC, qn_ctx_q, qn_ctx_k, qbuf, kbuf, vbuf, 256,  0,   0, Wp, 10000.f);
  qkpost_kernel<<<4096, 256, 0, stream>>>(qkvI, qn_in_q,  qn_in_k,  qbuf, kbuf, vbuf, 1024, 256, 1, Wp, 100.f);

  attn_kernel<<<dim3(64, 10), 256, 0, stream>>>(qbuf, kbuf, vbuf, am, attnC, attnI);

  gemm_bf16<EPI_GATE><<<dim3(32, 8), 256, GEMM_LDS, stream>>>(attnI, wt_po_in,  po_in_b,  outX, nullptr, x,   mod_i + 2 * 1024, 4096, 1024, 1024, 1024);
  gemm_bf16<EPI_GATE><<<dim3(8,  8), 256, GEMM_LDS, stream>>>(attnC, wt_po_ctx, po_ctx_b, outC, nullptr, ctx, mod_c + 2 * 1024, 1024, 1024, 1024, 256);

  modnorm_kernel<<<4096, 256, 0, stream>>>(outX, norm2_in,  mod_i, 3, 4, xm2, 1024);
  modnorm_kernel<<<1024, 256, 0, stream>>>(outC, norm2_ctx, mod_c, 3, 4, cm2, 256);

  gemm_bf16<EPI_GELU_BF16><<<dim3(32, 32), 256, GEMM_LDS, stream>>>(xm2, wt_mlp_i1, mlp_in_b1,  nullptr, hIn, nullptr, nullptr, 4096, 4096, 1024, 1024);
  gemm_bf16<EPI_GATE>     <<<dim3(32, 8),  256, GEMM_LDS, stream>>>(hIn, wt_mlp_i2, mlp_in_b2,  outX, nullptr, outX, mod_i + 5 * 1024, 4096, 1024, 4096, 1024);

  gemm_bf16<EPI_GELU_BF16><<<dim3(8, 32), 256, GEMM_LDS, stream>>>(cm2, wt_mlp_c1, mlp_ctx_b1, nullptr, hC, nullptr, nullptr, 1024, 4096, 1024, 256);
  gemm_bf16<EPI_GATE>     <<<dim3(8, 8),  256, GEMM_LDS, stream>>>(hC,  wt_mlp_c2, mlp_ctx_b2, outC, nullptr, outC, mod_c + 5 * 1024, 1024, 1024, 4096, 256);
}
